// SmallTransformerClassifier_88656714925574
// MI455X (gfx1250) — matrix operand order repaired and checked by reading, not compiled, not run on hardware
//
#include <hip/hip_runtime.h>

// ---------------------------------------------------------------------------
// MI455X (gfx1250, wave32) GPT forward pass.
// bf16 WMMA (v_wmma_f32_16x16x32_bf16) for all GEMMs + flash attention.
// Round 3: TDM (tensor_load_to_lds) double-buffered staging of the shared
// A tile in the GEMM — all 4 waves of a block consume one DMA'd copy from
// LDS instead of 4x redundant global reads; synced with s_wait_tensorcnt.
// ---------------------------------------------------------------------------

typedef __attribute__((ext_vector_type(16))) __bf16 v16bf;
typedef __attribute__((ext_vector_type(8)))  float  v8f;
typedef __attribute__((ext_vector_type(4)))  unsigned int v4u;
typedef __attribute__((ext_vector_type(8)))  int    v8i;
typedef __attribute__((ext_vector_type(4)))  int    v4i;

#define WMMA_BF16(a, b, c) \
  __builtin_amdgcn_wmma_f32_16x16x32_bf16(false, (a), false, (b), (short)0, (c), false, false)

// ---- Tensor Data Mover: DMA a [tile_h x tile_w] bf16 tile into LDS --------
// D# layout per CDNA5 ISA. Groups 2/3 are zero (2-D
// tile); workgroup_mask=0 (not dispatched in a cluster). Completion tracked
// by TENSORcnt (s_wait_tensorcnt).
__device__ __forceinline__ void tdm_load_bf16_tile(const __bf16* gsrc,
                                                   unsigned lds_byte_off,
                                                   unsigned tensor_w,
                                                   unsigned tensor_h,
                                                   unsigned tile_w,
                                                   unsigned tile_h,
                                                   unsigned row_stride) {
  unsigned long long ga = (unsigned long long)(uintptr_t)gsrc;
  v4u g0;
  g0[0] = 1u;                                                  // count=1, user mode
  g0[1] = lds_byte_off;                                        // LDS dest (bytes)
  g0[2] = (unsigned)ga;                                        // global addr [31:0]
  g0[3] = (unsigned)((ga >> 32) & 0x01FFFFFFu) | (2u << 30);   // addr[56:32] | type=2
  v8i g1;
  g1[0] = (int)(1u << 16);                                     // wg_mask=0, data_size=1 (2B)
  g1[1] = (int)((tensor_w & 0xFFFFu) << 16);                   // tensor_dim0[15:0]
  g1[2] = (int)((tensor_w >> 16) | ((tensor_h & 0xFFFFu) << 16));
  g1[3] = (int)((tensor_h >> 16) | ((tile_w & 0xFFFFu) << 16));// | tile_dim0
  g1[4] = (int)(tile_h & 0xFFFFu);                             // tile_dim1, tile_dim2=0
  g1[5] = (int)row_stride;                                     // tensor_dim0_stride[31:0]
  g1[6] = 0;
  g1[7] = 0;
  v4i z = {0, 0, 0, 0};
#if defined(__clang_major__) && __clang_major__ >= 23
  v8i z8 = {0, 0, 0, 0, 0, 0, 0, 0};
  __builtin_amdgcn_tensor_load_to_lds(g0, g1, z, z, z8, 0);
#else
  __builtin_amdgcn_tensor_load_to_lds(g0, g1, z, z, 0);
#endif
}

// ---- DPP row-of-16 reductions (wave32: two independent 16-lane halves) ----
template <int CTRL>
__device__ __forceinline__ float dpp_mov(float x) {
  int r = __builtin_amdgcn_update_dpp(0, __float_as_int(x), CTRL, 0xf, 0xf, true);
  return __int_as_float(r);
}
__device__ __forceinline__ float half16_max(float x) {
  x = fmaxf(x, dpp_mov<0x128>(x));   // ROW_ROR:8
  x = fmaxf(x, dpp_mov<0x124>(x));   // ROW_ROR:4
  x = fmaxf(x, dpp_mov<0x122>(x));   // ROW_ROR:2
  x = fmaxf(x, dpp_mov<0x121>(x));   // ROW_ROR:1
  return x;
}
__device__ __forceinline__ float half16_sum(float x) {
  x += dpp_mov<0x128>(x);
  x += dpp_mov<0x124>(x);
  x += dpp_mov<0x122>(x);
  x += dpp_mov<0x121>(x);
  return x;
}

// ---------------------------------------------------------------------------
// Embedding: x[t, d] = tok_emb[ids[t]][d] + pos_emb[t % S][d]   (fp32 stream)
// ---------------------------------------------------------------------------
__global__ void embed_kernel(const int* __restrict__ ids,
                             const float* __restrict__ tok,
                             const float* __restrict__ pos,
                             float* __restrict__ x, int S, int D) {
  int t = blockIdx.x;                     // 0 .. B*S-1
  int s = t % S;
  const float* te = tok + (size_t)ids[t] * D;
  const float* pe = pos + (size_t)s * D;
  float* xr = x + (size_t)t * D;
  for (int d = threadIdx.x; d < D; d += blockDim.x) xr[d] = te[d] + pe[d];
}

// ---------------------------------------------------------------------------
// LayerNorm over D, output bf16 (feeds WMMA A operand)
// ---------------------------------------------------------------------------
__global__ void layernorm_kernel(const float* __restrict__ x,
                                 const float* __restrict__ w,
                                 const float* __restrict__ bia,
                                 __bf16* __restrict__ out, int D) {
  int row = blockIdx.x;
  const float* xr = x + (size_t)row * D;
  float s = 0.f, s2 = 0.f;
  for (int d = threadIdx.x; d < D; d += blockDim.x) {
    float v = xr[d]; s += v; s2 += v * v;
  }
  __shared__ float sb[256], sq[256];
  sb[threadIdx.x] = s; sq[threadIdx.x] = s2;
  __syncthreads();
  for (int off = blockDim.x >> 1; off > 0; off >>= 1) {
    if ((int)threadIdx.x < off) { sb[threadIdx.x] += sb[threadIdx.x + off];
                                  sq[threadIdx.x] += sq[threadIdx.x + off]; }
    __syncthreads();
  }
  float mu  = sb[0] / D;
  float var = sq[0] / D - mu * mu;
  float rst = rsqrtf(var + 1e-5f);
  __bf16* orow = out + (size_t)row * D;
  for (int d = threadIdx.x; d < D; d += blockDim.x)
    orow[d] = (__bf16)((xr[d] - mu) * rst * w[d] + bia[d]);
}

// Final LN applied only to the last token of each batch; builds a zero-padded
// 16 x D bf16 tile so the vocab head can run as one WMMA GEMM with M=16.
__global__ void ln_head_kernel(const float* __restrict__ x,
                               const float* __restrict__ w,
                               const float* __restrict__ bia,
                               __bf16* __restrict__ out, int B, int S, int D) {
  int i = blockIdx.x;                     // 0..15 (padded M tile)
  __bf16* orow = out + (size_t)i * D;
  if (i >= B) {
    for (int d = threadIdx.x; d < D; d += blockDim.x) orow[d] = (__bf16)0.f;
    return;
  }
  const float* xr = x + ((size_t)i * S + (S - 1)) * D;
  float s = 0.f, s2 = 0.f;
  for (int d = threadIdx.x; d < D; d += blockDim.x) {
    float v = xr[d]; s += v; s2 += v * v;
  }
  __shared__ float sb[256], sq[256];
  sb[threadIdx.x] = s; sq[threadIdx.x] = s2;
  __syncthreads();
  for (int off = blockDim.x >> 1; off > 0; off >>= 1) {
    if ((int)threadIdx.x < off) { sb[threadIdx.x] += sb[threadIdx.x + off];
                                  sq[threadIdx.x] += sq[threadIdx.x + off]; }
    __syncthreads();
  }
  float mu  = sb[0] / D;
  float var = sq[0] / D - mu * mu;
  float rst = rsqrtf(var + 1e-5f);
  for (int d = threadIdx.x; d < D; d += blockDim.x)
    orow[d] = (__bf16)((xr[d] - mu) * rst * w[d] + bia[d]);
}

// ---------------------------------------------------------------------------
// WMMA GEMM:  C[M,N] = A[M,K](bf16) @ W[N,K]^T (fp32 -> bf16 on the fly)
// The block's 16-row A tile is DMA'd into LDS by the TDM (wave 0 issues
// tensor_load_to_lds, double-buffered 16x64 chunks, s_wait_tensorcnt +
// workgroup barrier), then all 4 waves read fragments from LDS. Weights
// stream straight from global (prefetched; NT for the zero-reuse head).
// EPI: 0 = +bias -> bf16 | 1 = +bias+GELU -> bf16
//      2 = +bias+resid -> fp32 | 3 = head: no bias, fp32, rows < Mvalid
// Note: grid is sized so every N-tile is in range (no early return; the
// k-loop contains barriers).
// ---------------------------------------------------------------------------
template <int EPI>
__global__ __launch_bounds__(128, 2)
void gemm_wmma_kernel(const __bf16* __restrict__ A,
                      const float* __restrict__ W,
                      const float* __restrict__ bias,
                      const float* __restrict__ resid,
                      float* __restrict__ outF,
                      __bf16* __restrict__ outB,
                      int M, int N, int K, int Mvalid) {
  // Only shared object in this kernel => LDS offset 0 (buf 1 at +2KB).
  __shared__ __bf16 As[2][16 * 64];
  int lane = threadIdx.x & 31;
  int wave = threadIdx.x >> 5;
  int half = lane >> 4;                   // wave32 half
  int ln   = lane & 15;
  int tN = (blockIdx.x * 4 + wave) * 2;   // first of two N-tiles
  int tM = blockIdx.y;

  const float* W0 = W + (size_t)(tN * 16 + ln) * K;   // weight rows = out cols
  const float* W1 = W0 + (size_t)16 * K;
  const __bf16* Abase = A + (size_t)(tM * 16) * K;    // block's 16-row A tile
  int nchunks = K / 64;

  if (wave == 0) {                        // kick chunk 0; EXEC-independent DMA
    tdm_load_bf16_tile(Abase, 0u, (unsigned)K, 16u, 64u, 16u, (unsigned)K);
    __builtin_amdgcn_s_wait_tensorcnt(0);
  }
  __syncthreads();

  v8f acc0 = {}, acc1 = {};
  for (int c = 0; c < nchunks; ++c) {
    int buf = c & 1;
    if (wave == 0 && (c + 1) < nchunks)   // async: DMA next chunk while computing
      tdm_load_bf16_tile(Abase + (size_t)(c + 1) * 64,
                         (unsigned)((buf ^ 1) * (16 * 64 * 2)),
                         (unsigned)K, 16u, 64u, 16u, (unsigned)K);
    const __bf16* Ac = &As[buf][0];
    int k0g = c * 64;
#pragma unroll
    for (int kk = 0; kk < 64; kk += 32) {
      int kb = k0g + kk + half * 16;      // B frag: halves take contiguous K 0-15/16-31
      if (kk == 0 && k0g + 128 < K) {     // stream weights: prefetch 2 chunks ahead
        __builtin_prefetch((const void*)(W0 + kb + 128), 0, 3);
        __builtin_prefetch((const void*)(W1 + kb + 128), 0, 3);
      }
      v16bf a, b0, b1;
#pragma unroll
      for (int j = 0; j < 8; ++j) {       // A frag from LDS (broadcast to 4 waves)
        a[j]     = Ac[ln * 64 + kk + half * 16 + j];
        a[j + 8] = Ac[ln * 64 + kk + half * 16 + 8 + j];
      }
#pragma unroll
      for (int j = 0; j < 16; ++j) {
        float w0v = (EPI == 3) ? __builtin_nontemporal_load(W0 + kb + j) : W0[kb + j];
        float w1v = (EPI == 3) ? __builtin_nontemporal_load(W1 + kb + j) : W1[kb + j];
        b0[j] = (__bf16)w0v;
        b1[j] = (__bf16)w1v;
      }
      acc0 = WMMA_BF16(a, b0, acc0);
      acc1 = WMMA_BF16(a, b1, acc1);
    }
    if (wave == 0 && (c + 1) < nchunks)
      __builtin_amdgcn_s_wait_tensorcnt(0);
    __syncthreads();                      // publish next buffer / retire this one
  }

#pragma unroll
  for (int t = 0; t < 2; ++t) {
    v8f acc = t ? acc1 : acc0;
    int col = (tN + t) * 16 + ln;         // C layout: N = lane&15
    float bv = (EPI == 3) ? 0.f : bias[col];
#pragma unroll
    for (int r = 0; r < 8; ++r) {         // C layout: M = r + half*8
      int row = tM * 16 + r + half * 8;
      float v = acc[r] + bv;
      if (EPI == 0) {
        outB[(size_t)row * N + col] = (__bf16)v;
      } else if (EPI == 1) {
        float g = 0.5f * v * (1.f + erff(v * 0.70710678118654752f));
        outB[(size_t)row * N + col] = (__bf16)g;
      } else if (EPI == 2) {
        outF[(size_t)row * N + col] = resid[(size_t)row * N + col] + v;
      } else {
        if (row < Mvalid) outF[(size_t)row * N + col] = v;
      }
    }
  }
}

// ---------------------------------------------------------------------------
// Flash-style causal attention. One wave per (batch, head, 16-row q-tile).
// QK^T and P*V are WMMA; online softmax reduces rows across the 16-lane
// halves with DPP rotates; P is staged through 1KB LDS (C-layout -> A-layout).
// ---------------------------------------------------------------------------
__global__ __launch_bounds__(32, 1)
void attn_kernel(const __bf16* __restrict__ qb,
                 const __bf16* __restrict__ kb,
                 const __bf16* __restrict__ vb,
                 __bf16* __restrict__ yb, int S, int D) {
  const int HD = 64;
  int lane = threadIdx.x;                 // block = 1 wave
  int half = lane >> 4, ln = lane & 15;
  int qt = blockIdx.x, h = blockIdx.y, b = blockIdx.z;
  size_t base = (size_t)b * S * D + (size_t)h * HD;

  // Q A-fragments for head-dim chunks [0,32) and [32,64)
  const __bf16* Qr = qb + base + (size_t)(qt * 16 + ln) * D;
  v16bf qf0, qf1;
#pragma unroll
  for (int j = 0; j < 8; ++j) {
    qf0[j]     = Qr[half * 16 + j];
    qf0[j + 8] = Qr[half * 16 + 8 + j];
    qf1[j]     = Qr[32 + half * 16 + j];
    qf1[j + 8] = Qr[32 + half * 16 + 8 + j];
  }

  float m[8], lsum[8];
  v8f acc[4] = {};
#pragma unroll
  for (int r = 0; r < 8; ++r) { m[r] = -1e30f; lsum[r] = 0.f; }

  __shared__ __bf16 pshare[16][32];
  const float scale = 0.125f;             // 1/sqrt(64)
  int kend = qt * 16 + 16;

  for (int kc = 0; kc < kend; kc += 32) {
    // ---- scores for keys [kc, kc+32): two 16x16 C tiles -------------------
    v8f s0 = {}, s1 = {};
    const __bf16* K0 = kb + base + (size_t)(kc + ln) * D;        // key kc+ln
    const __bf16* K1 = K0 + (size_t)16 * D;                      // key kc+16+ln
    v16bf bf;
#pragma unroll
    for (int j = 0; j < 16; ++j) bf[j] = K0[half * 16 + j];      // d 0..31
    s0 = WMMA_BF16(qf0, bf, s0);
#pragma unroll
    for (int j = 0; j < 16; ++j) bf[j] = K0[32 + half * 16 + j]; // d 32..63
    s0 = WMMA_BF16(qf1, bf, s0);
#pragma unroll
    for (int j = 0; j < 16; ++j) bf[j] = K1[half * 16 + j];
    s1 = WMMA_BF16(qf0, bf, s1);
#pragma unroll
    for (int j = 0; j < 16; ++j) bf[j] = K1[32 + half * 16 + j];
    s1 = WMMA_BF16(qf1, bf, s1);

    // ---- scale, causal mask, online softmax (rows live in lane halves) ----
    int col0 = kc + ln, col1 = col0 + 16;
#pragma unroll
    for (int r = 0; r < 8; ++r) {
      int qrow = qt * 16 + r + half * 8;
      float x0 = (col0 <= qrow) ? s0[r] * scale : -1e30f;
      float x1 = (col1 <= qrow) ? s1[r] * scale : -1e30f;
      float mx   = half16_max(fmaxf(x0, x1));
      float mnew = fmaxf(m[r], mx);
      float corr = __expf(m[r] - mnew);
      float p0 = __expf(x0 - mnew);
      float p1 = __expf(x1 - mnew);
      float rs = half16_sum(p0 + p1);
      lsum[r] = lsum[r] * corr + rs;
      m[r] = mnew;
#pragma unroll
      for (int dt = 0; dt < 4; ++dt) acc[dt][r] *= corr;
      pshare[r + half * 8][ln]      = (__bf16)p0;   // C-layout spill of P
      pshare[r + half * 8][ln + 16] = (__bf16)p1;
    }
    __syncthreads();

    // P as 16x32 A fragment
    v16bf pa;
#pragma unroll
    for (int j = 0; j < 8; ++j) {
      pa[j]     = pshare[ln][half * 16 + j];
      pa[j + 8] = pshare[ln][half * 16 + 8 + j];
    }
    __syncthreads();

    // ---- P @ V over this 32-key chunk, 4 head-dim column tiles ------------
#pragma unroll
    for (int dt = 0; dt < 4; ++dt) {
      const __bf16* V0 = vb + base + dt * 16 + ln;  // out column d = dt*16+ln
      v16bf bv;
#pragma unroll
      for (int j = 0; j < 16; ++j)
        bv[j] = V0[(size_t)(kc + half * 16 + j) * D];
      acc[dt] = WMMA_BF16(pa, bv, acc[dt]);
    }
  }

  // ---- normalize and write y (bf16, feeds O-projection GEMM) --------------
#pragma unroll
  for (int r = 0; r < 8; ++r) {
    int row = qt * 16 + r + half * 8;
    float inv = 1.f / lsum[r];
#pragma unroll
    for (int dt = 0; dt < 4; ++dt)
      yb[base + (size_t)row * D + dt * 16 + ln] = (__bf16)(acc[dt][r] * inv);
  }
}

// ---------------------------------------------------------------------------
// Orchestration
// ---------------------------------------------------------------------------
extern "C" void kernel_launch(void* const* d_in, const int* in_sizes, int n_in,
                              void* d_out, int out_size, void* d_ws, size_t ws_size,
                              hipStream_t stream) {
  (void)in_sizes; (void)n_in; (void)out_size; (void)ws_size;
  const int B = 2, S = 1024, D = 1024, H = 16, L = 4, FF = 4096, V = 32000;
  const int MT = B * S;                   // 2048 token rows

  const int*   ids  = (const int*)d_in[0];
  const float* tok  = (const float*)d_in[1];
  const float* pos  = (const float*)d_in[2];
  const float* ln1w = (const float*)d_in[3],  *ln1b = (const float*)d_in[4];
  const float* ln2w = (const float*)d_in[5],  *ln2b = (const float*)d_in[6];
  const float* qw   = (const float*)d_in[7],  *qbi  = (const float*)d_in[8];
  const float* kw   = (const float*)d_in[9],  *kbi  = (const float*)d_in[10];
  const float* vw   = (const float*)d_in[11], *vbi  = (const float*)d_in[12];
  const float* ow   = (const float*)d_in[13], *obi  = (const float*)d_in[14];
  const float* f1w  = (const float*)d_in[15], *f1b  = (const float*)d_in[16];
  const float* f2w  = (const float*)d_in[17], *f2b  = (const float*)d_in[18];
  const float* lnfw = (const float*)d_in[19], *lnfb = (const float*)d_in[20];
  const float* hw   = (const float*)d_in[21];

  // Workspace carve-out (~44 MB)
  char* p = (char*)d_ws;
  float*  xf  = (float*)p;   p += (size_t)MT * D * sizeof(float);   // residual stream
  __bf16* hb  = (__bf16*)p;  p += (size_t)MT * D * 2;               // LN output
  __bf16* qb  = (__bf16*)p;  p += (size_t)MT * D * 2;
  __bf16* kbq = (__bf16*)p;  p += (size_t)MT * D * 2;
  __bf16* vbq = (__bf16*)p;  p += (size_t)MT * D * 2;
  __bf16* yb  = (__bf16*)p;  p += (size_t)MT * D * 2;               // attn out
  __bf16* ffb = (__bf16*)p;  p += (size_t)MT * FF * 2;              // fc1+gelu out
  __bf16* hin = (__bf16*)p;  p += (size_t)16 * D * 2;               // padded head in

  embed_kernel<<<dim3(MT), dim3(256), 0, stream>>>(ids, tok, pos, xf, S, D);

  dim3 gP(D / 128, MT / 16);              // 128 output cols per block
  dim3 gF(FF / 128, MT / 16);
  for (int i = 0; i < L; ++i) {
    layernorm_kernel<<<MT, 256, 0, stream>>>(xf, ln1w + i * D, ln1b + i * D, hb, D);
    gemm_wmma_kernel<0><<<gP, 128, 0, stream>>>(hb, qw + (size_t)i * D * D, qbi + i * D,
                                                nullptr, nullptr, qb, MT, D, D, MT);
    gemm_wmma_kernel<0><<<gP, 128, 0, stream>>>(hb, kw + (size_t)i * D * D, kbi + i * D,
                                                nullptr, nullptr, kbq, MT, D, D, MT);
    gemm_wmma_kernel<0><<<gP, 128, 0, stream>>>(hb, vw + (size_t)i * D * D, vbi + i * D,
                                                nullptr, nullptr, vbq, MT, D, D, MT);
    attn_kernel<<<dim3(S / 16, H, B), 32, 0, stream>>>(qb, kbq, vbq, yb, S, D);
    gemm_wmma_kernel<2><<<gP, 128, 0, stream>>>(yb, ow + (size_t)i * D * D, obi + i * D,
                                                xf, xf, nullptr, MT, D, D, MT);
    layernorm_kernel<<<MT, 256, 0, stream>>>(xf, ln2w + i * D, ln2b + i * D, hb, D);
    gemm_wmma_kernel<1><<<gF, 128, 0, stream>>>(hb, f1w + (size_t)i * FF * D, f1b + i * FF,
                                                nullptr, nullptr, ffb, MT, FF, D, MT);
    gemm_wmma_kernel<2><<<gP, 128, 0, stream>>>(ffb, f2w + (size_t)i * D * FF, f2b + i * D,
                                                xf, xf, nullptr, MT, D, FF, MT);
  }

  ln_head_kernel<<<16, 256, 0, stream>>>(xf, lnfw, lnfb, hin, B, S, D);
  gemm_wmma_kernel<3><<<dim3(V / 128, 1), 128, 0, stream>>>(hin, hw, nullptr, nullptr,
                                                            (float*)d_out, nullptr,
                                                            16, V, D, B);
}
